// ExpressionModel_56813827391507
// MI455X (gfx1250) — matrix operand order repaired and checked by reading, not compiled, not run on hardware
//
#include <hip/hip_runtime.h>

#define B_  2
#define L_  2048
#define C_  1024
#define H_  16
#define D_  64
#define L2_ 512
#define TD_ 768
#define FF_ 4096

typedef __bf16 bf16;
typedef __attribute__((ext_vector_type(16))) __bf16 v16bf;
typedef __attribute__((ext_vector_type(8)))  float  v8f;
typedef int v4i_ __attribute__((vector_size(16)));

union BfPair { unsigned u; bf16 h[2]; };
union Frag16 { unsigned u[8]; v16bf v; };

#define AS1 __attribute__((address_space(1)))
#define AS3 __attribute__((address_space(3)))

// ---- CDNA5 async global->LDS staging (ASYNCcnt path), with portable fallback ----
static __device__ __forceinline__ void async_b128(const void* g, void* l) {
#if __has_builtin(__builtin_amdgcn_global_load_async_to_lds_b128)
    __builtin_amdgcn_global_load_async_to_lds_b128(
        (AS1 v4i_*)(uintptr_t)g,
        (AS3 v4i_*)(unsigned)(uintptr_t)l, 0, 0);
#else
    *reinterpret_cast<uint4*>(l) = *reinterpret_cast<const uint4*>(g);
#endif
}
static __device__ __forceinline__ void wait_async() {
#if __has_builtin(__builtin_amdgcn_global_load_async_to_lds_b128)
#if __has_builtin(__builtin_amdgcn_s_wait_asynccnt)
    __builtin_amdgcn_s_wait_asynccnt(0);
#else
    asm volatile("s_wait_asynccnt 0x0" ::: "memory");
#endif
#endif
}

static __device__ __forceinline__ float sigmoidf_(float x) { return 1.f / (1.f + __expf(-x)); }

// ---------------- elementwise kernels ----------------

__global__ void k_f32_to_bf16(const float* __restrict__ in, bf16* __restrict__ out, int n) {
    int i = blockIdx.x * blockDim.x + threadIdx.x;
    if (i < n) out[i] = (bf16)in[i];
}

// convert + transpose: in [K][N] f32 -> out [N][K] bf16   (weights -> GEMM B operand)
__global__ void k_cvt_t(const float* __restrict__ in, bf16* __restrict__ out, int K, int N) {
    int i = blockIdx.x * blockDim.x + threadIdx.x;
    if (i >= K * N) return;
    int n = i % N, k = i / N;
    out[(size_t)n * K + k] = (bf16)in[i];
}

__global__ void k_silu(const float* __restrict__ in, float* __restrict__ out, int n) {
    int i = blockIdx.x * blockDim.x + threadIdx.x;
    if (i < n) { float v = in[i]; out[i] = v * sigmoidf_(v); }
}

// mods[b, col] = bias[col] + sum_k silu_t[b,k] * W[k, col]   (2 x 6144, K=1024)
__global__ void k_adaln(const float* __restrict__ s, const float* __restrict__ W,
                        const float* __restrict__ bias, float* __restrict__ mods) {
    int i = blockIdx.x * blockDim.x + threadIdx.x;   // 0 .. 12287
    int col = i % (6 * C_);
    int b   = i / (6 * C_);
    float acc = bias[col];
    const float* sr = s + b * C_;
    for (int k = 0; k < C_; k++) acc += sr[k] * W[k * (6 * C_) + col];
    mods[i] = acc;
}

// out_bf16 = rmsnorm(x)*w*(1+scale) + shift ; scale_off/shift_off < 0 -> disabled
__global__ __launch_bounds__(256) void k_rmsnorm_mod(
        const float* __restrict__ x, const float* __restrict__ w,
        const float* __restrict__ mods, int scale_off, int shift_off,
        bf16* __restrict__ out) {
    int row = blockIdx.x;          // B*L rows
    int b   = row >> 11;
    int tid = threadIdx.x;
    const float* xr = x + (size_t)row * C_;
    float ss = 0.f;
    for (int c = tid; c < C_; c += 256) { float v = xr[c]; ss += v * v; }
#pragma unroll
    for (int m = 16; m >= 1; m >>= 1) ss += __shfl_xor(ss, m, 32);
    __shared__ float red[8];
    if ((tid & 31) == 0) red[tid >> 5] = ss;
    __syncthreads();
    float tot = 0.f;
#pragma unroll
    for (int i = 0; i < 8; i++) tot += red[i];
    float rs = rsqrtf(tot * (1.f / C_) + 1e-6f);
    bf16* orow = out + (size_t)row * C_;
    const float* mrow = mods + b * (6 * C_);
    for (int c = tid; c < C_; c += 256) {
        float n = xr[c] * rs * w[c];
        if (scale_off >= 0) n *= (1.f + mrow[scale_off + c]);
        if (shift_off >= 0) n += mrow[shift_off + c];
        orow[c] = (bf16)n;
    }
}

// out = x + gate * y
__global__ void k_residual(const float* __restrict__ x, const float* __restrict__ y,
                           const float* __restrict__ mods, int gate_off,
                           float* __restrict__ out, int n) {
    int i = blockIdx.x * blockDim.x + threadIdx.x;
    if (i >= n) return;
    int c = i & (C_ - 1);
    int b = i >> 21;
    float g = (gate_off >= 0) ? mods[b * (6 * C_) + gate_off + c] : 1.f;
    out[i] = x[i] + g * y[i];
}

__global__ void k_swiglu(const float* __restrict__ g, const float* __restrict__ u,
                         bf16* __restrict__ h, int n) {
    int i = blockIdx.x * blockDim.x + threadIdx.x;
    if (i < n) { float gv = g[i]; h[i] = (bf16)(gv * sigmoidf_(gv) * u[i]); }
}

// qkv f32 [B*L][3C] -> rope(q), rope(k), v  as bf16 [B*H][L][D]
__global__ void k_rope_pack(const float* __restrict__ qkv,
                            const float* __restrict__ fc, const float* __restrict__ fs,
                            bf16* __restrict__ Qh, bf16* __restrict__ Kh, bf16* __restrict__ Vh) {
    int i  = blockIdx.x * blockDim.x + threadIdx.x;  // 2^21
    int pr = i & 31;
    int h  = (i >> 5) & 15;
    int l  = (i >> 9) & 2047;
    int b  = i >> 20;
    const float* row = qkv + (size_t)(b * L_ + l) * (3 * C_);
    int base = h * D_ + 2 * pr;
    float c = fc[l * 32 + pr], s = fs[l * 32 + pr];
    int o = ((b * H_ + h) * L_ + l) * D_ + 2 * pr;
    float xr = row[base], xi = row[base + 1];
    Qh[o]     = (bf16)(xr * c - xi * s);
    Qh[o + 1] = (bf16)(xr * s + xi * c);
    xr = row[C_ + base]; xi = row[C_ + base + 1];
    Kh[o]     = (bf16)(xr * c - xi * s);
    Kh[o + 1] = (bf16)(xr * s + xi * c);
    Vh[o]     = (bf16)row[2 * C_ + base];
    Vh[o + 1] = (bf16)row[2 * C_ + base + 1];
}

// q f32 [B*L][C] -> bf16 [B*H][L][D]
__global__ void k_pack_q(const float* __restrict__ q, bf16* __restrict__ Qh) {
    int i = blockIdx.x * blockDim.x + threadIdx.x;   // 2^22
    int d = i & 63; int h = (i >> 6) & 15; int l = (i >> 10) & 2047; int b = i >> 21;
    Qh[((b * H_ + h) * L_ + l) * D_ + d] = (bf16)q[i];
}

// kv f32 [B*L2][2C] -> k,v bf16 [B*H][L2][D]
__global__ void k_pack_kv(const float* __restrict__ kv, bf16* __restrict__ Kh, bf16* __restrict__ Vh) {
    int i = blockIdx.x * blockDim.x + threadIdx.x;   // 2^20
    int d = i & 63; int h = (i >> 6) & 15; int l = (i >> 10) & 511; int b = i >> 19;
    const float* row = kv + (size_t)(b * L2_ + l) * (2 * C_);
    int o = ((b * H_ + h) * L2_ + l) * D_ + d;
    Kh[o] = (bf16)row[h * D_ + d];
    Vh[o] = (bf16)row[C_ + h * D_ + d];
}

// ---------------- bf16 WMMA GEMM: C[MxN] = A[MxK] * Bt[NxK]^T ----------------
// 256 threads = 8 waves, block tile 128x128, wave tile 32x64, K step 32.
// Double-buffered LDS staged with async global->LDS loads.
__global__ __launch_bounds__(256) void k_gemm_bf16(
        const bf16* __restrict__ A, const bf16* __restrict__ Bt,
        float* __restrict__ C, int M, int N, int K) {
    __shared__ __align__(16) bf16 sA[2][128 * 32];   // [m][k]
    __shared__ __align__(16) bf16 sB[2][128 * 32];   // [n][k]
    const int tid  = threadIdx.x;
    const int lane = tid & 31, wave = tid >> 5;
    const int lm = lane & 15, half = lane >> 4;
    const int m0 = blockIdx.y * 128, n0 = blockIdx.x * 128;
    const int wm = (wave >> 1) * 32, wn = (wave & 1) * 64;
    const int r = tid >> 1, cc = (tid & 1) * 16;     // 16 elems (32B) per thread per tile
    const bf16* gA = A  + (size_t)(m0 + r) * K + cc;
    const bf16* gB = Bt + (size_t)(n0 + r) * K + cc;

    v8f acc[2][4];
#pragma unroll
    for (int i = 0; i < 2; i++)
#pragma unroll
        for (int j = 0; j < 4; j++)
#pragma unroll
            for (int e = 0; e < 8; e++) acc[i][j][e] = 0.f;

    auto stage = [&](int k0, int buf) {
        async_b128(gA + k0,     &sA[buf][r * 32 + cc]);
        async_b128(gA + k0 + 8, &sA[buf][r * 32 + cc + 8]);
        async_b128(gB + k0,     &sB[buf][r * 32 + cc]);
        async_b128(gB + k0 + 8, &sB[buf][r * 32 + cc + 8]);
    };

    const int nk = K >> 5;
    stage(0, 0);
    for (int i = 0; i < nk; i++) {
        wait_async();
        __syncthreads();
        if (i + 1 < nk) stage((i + 1) << 5, (i + 1) & 1);
        const bf16* aT = sA[i & 1];
        const bf16* bT = sB[i & 1];
        v16bf af[2], bfv[4];
#pragma unroll
        for (int t = 0; t < 2; t++) {
            const unsigned* arow = reinterpret_cast<const unsigned*>(aT + (wm + t * 16 + lm) * 32);
            Frag16 f;
#pragma unroll
            for (int jj = 0; jj < 8; jj++) {
                int kk = (half ? 16 : 0) + 2 * jj;
                f.u[jj] = arow[kk >> 1];
            }
            af[t] = f.v;
        }
#pragma unroll
        for (int t = 0; t < 4; t++) {
            const unsigned* brow = reinterpret_cast<const unsigned*>(bT + (wn + t * 16 + lm) * 32);
            Frag16 f;
#pragma unroll
            for (int jj = 0; jj < 8; jj++) f.u[jj] = brow[(half ? 8 : 0) + jj];
            bfv[t] = f.v;
        }
#pragma unroll
        for (int i2 = 0; i2 < 2; i2++)
#pragma unroll
            for (int j = 0; j < 4; j++)
                acc[i2][j] = __builtin_amdgcn_wmma_f32_16x16x32_bf16(
                    false, af[i2], false, bfv[j], (short)0, acc[i2][j], false, false);
    }
    // epilogue: 32-bit offsets (SGPR base + VGPR offset addressing)
#pragma unroll
    for (int i = 0; i < 2; i++)
#pragma unroll
        for (int j = 0; j < 4; j++) {
            int nn = n0 + wn + j * 16 + lm;
            int mb = m0 + wm + i * 16 + (half ? 8 : 0);
            int base = mb * N + nn;
#pragma unroll
            for (int e = 0; e < 8; e++) C[base + e * N] = acc[i][j][e];
        }
}

// ---------------- flash attention (bf16 WMMA, online softmax) ----------------
// Q/K/V: [B*H][L][64] bf16. O: [B*L][H*64] bf16. 128 threads = 4 waves; wave = 16 q rows.
// Computes S^T = K*Q^T (per-lane softmax stats); O^T = V^T * P^T. Async-staged K/V.
__global__ __launch_bounds__(128) void k_flash(
        const bf16* __restrict__ Qh, const bf16* __restrict__ Kh, const bf16* __restrict__ Vh,
        bf16* __restrict__ O, int Lq, int Lk, float scale) {
    __shared__ __align__(16) bf16 sK[2][32 * 64];
    __shared__ __align__(16) bf16 sV[2][32 * 64];
    const int tid  = threadIdx.x;
    const int lane = tid & 31, wave = tid >> 5;
    const int lm = lane & 15, half = lane >> 4;
    const int bh = blockIdx.y, b = bh >> 4, h = bh & 15;
    const int q0 = blockIdx.x * 64 + wave * 16;

    // Q^T as B-fragments (n = q, k-dim = d)
    const bf16* qrow = Qh + ((size_t)bh * Lq + q0 + lm) * D_;
    v16bf qf[2];
#pragma unroll
    for (int f = 0; f < 2; f++) {
        Frag16 fr;
#pragma unroll
        for (int jj = 0; jj < 8; jj++) {
            int d = f * 32 + (half ? 16 : 0) + 2 * jj;
            fr.u[jj] = *reinterpret_cast<const unsigned*>(qrow + d);
        }
        qf[f] = fr.v;
    }
    v8f ot[4];
#pragma unroll
    for (int s = 0; s < 4; s++)
#pragma unroll
        for (int e = 0; e < 8; e++) ot[s][e] = 0.f;
    float Mx = -1e30f, Sum = 0.f;

    const int sr = tid >> 2, sc = (tid & 3) * 16;    // 16 elems per thread per matrix
    const bf16* gK = Kh + (size_t)bh * Lk * D_ + sr * D_ + sc;
    const bf16* gV = Vh + (size_t)bh * Lk * D_ + sr * D_ + sc;
    auto stage = [&](int ch, int buf) {
        int go = ch * 32 * D_;
        async_b128(gK + go,     &sK[buf][sr * 64 + sc]);
        async_b128(gK + go + 8, &sK[buf][sr * 64 + sc + 8]);
        async_b128(gV + go,     &sV[buf][sr * 64 + sc]);
        async_b128(gV + go + 8, &sV[buf][sr * 64 + sc + 8]);
    };

    const int nc = Lk >> 5;
    stage(0, 0);
    for (int i = 0; i < nc; i++) {
        wait_async();
        __syncthreads();
        if (i + 1 < nc) stage(i + 1, (i + 1) & 1);
        const bf16* cK = sK[i & 1];
        const bf16* cV = sV[i & 1];

        // S^T tiles: rows = keys (2 subtiles of 16), cols = 16 q
        float p[2][8];
        float cmax = -3.0e38f;
#pragma unroll
        for (int t = 0; t < 2; t++) {
            v8f cs;
#pragma unroll
            for (int e = 0; e < 8; e++) cs[e] = 0.f;
#pragma unroll
            for (int f = 0; f < 2; f++) {
                const unsigned* krow = reinterpret_cast<const unsigned*>(cK + (t * 16 + lm) * 64);
                Frag16 fr;
#pragma unroll
                for (int jj = 0; jj < 8; jj++) {
                    int d = f * 32 + (half ? 16 : 0) + 2 * jj;
                    fr.u[jj] = krow[d >> 1];
                }
                cs = __builtin_amdgcn_wmma_f32_16x16x32_bf16(
                    false, fr.v, false, qf[f], (short)0, cs, false, false);
            }
#pragma unroll
            for (int e = 0; e < 8; e++) { float sv = cs[e] * scale; p[t][e] = sv; cmax = fmaxf(cmax, sv); }
        }
        cmax = fmaxf(cmax, __shfl_xor(cmax, 16, 32));
        float nM = fmaxf(Mx, cmax);
        float corr = __expf(Mx - nM);
        float csum = 0.f;
#pragma unroll
        for (int t = 0; t < 2; t++)
#pragma unroll
            for (int e = 0; e < 8; e++) { p[t][e] = __expf(p[t][e] - nM); csum += p[t][e]; }
        csum += __shfl_xor(csum, 16, 32);
        Sum = Sum * corr + csum; Mx = nM;
#pragma unroll
        for (int s = 0; s < 4; s++)
#pragma unroll
            for (int e = 0; e < 8; e++) ot[s][e] *= corr;

        // C-layout P^T -> B-fragment (cross-half key exchange via shfl_xor 16)
        float oth[2][8];
#pragma unroll
        for (int t = 0; t < 2; t++)
#pragma unroll
            for (int e = 0; e < 8; e++) oth[t][e] = __shfl_xor(p[t][e], 16, 32);
        Frag16 pf;
#pragma unroll
        for (int jj = 0; jj < 8; jj++) {
            int i0 = 2 * jj, i1 = 2 * jj + 1;
            float x0, x1;
            if (half == 0) {   // keys 0..15
                x0 = (i0 < 8) ? p[0][i0] : oth[0][i0 - 8];
                x1 = (i1 < 8) ? p[0][i1] : oth[0][i1 - 8];
            } else {           // keys 16..31
                x0 = (i0 < 8) ? oth[1][i0] : p[1][i0 - 8];
                x1 = (i1 < 8) ? oth[1][i1] : p[1][i1 - 8];
            }
            BfPair bp; bp.h[0] = (bf16)x0; bp.h[1] = (bf16)x1;
            pf.u[jj] = bp.u;
        }
        // O^T += V^T * P^T (4 d-subtiles)
#pragma unroll
        for (int s = 0; s < 4; s++) {
            Frag16 vf;
            int d = s * 16 + lm;
#pragma unroll
            for (int jj = 0; jj < 8; jj++) {
                int k0 = (half ? 16 : 0) + 2 * jj;
                BfPair bp; bp.h[0] = cV[k0 * 64 + d]; bp.h[1] = cV[(k0 + 1) * 64 + d];
                vf.u[jj] = bp.u;
            }
            ot[s] = __builtin_amdgcn_wmma_f32_16x16x32_bf16(
                false, vf.v, false, pf.v, (short)0, ot[s], false, false);
        }
    }
    // epilogue: packed b128 stores (8 contiguous bf16 per lane per subtile)
    float inv = 1.f / Sum;
#pragma unroll
    for (int s = 0; s < 4; s++) {
        uint4 pk;
        unsigned* pu = reinterpret_cast<unsigned*>(&pk);
#pragma unroll
        for (int e = 0; e < 4; e++) {
            BfPair bp;
            bp.h[0] = (bf16)(ot[s][2 * e]     * inv);
            bp.h[1] = (bf16)(ot[s][2 * e + 1] * inv);
            pu[e] = bp.u;
        }
        int ob = ((b * Lq + q0 + lm) * H_ + h) * D_ + s * 16 + half * 8;
        *reinterpret_cast<uint4*>(O + ob) = pk;
    }
}

// ---------------- orchestration ----------------

extern "C" void kernel_launch(void* const* d_in, const int* in_sizes, int n_in,
                              void* d_out, int out_size, void* d_ws, size_t ws_size,
                              hipStream_t stream) {
    (void)in_sizes; (void)n_in; (void)out_size; (void)ws_size;
    const float* x     = (const float*)d_in[0];
    const float* t_mod = (const float*)d_in[1];
    const float* audio = (const float*)d_in[2];
    const float* fc    = (const float*)d_in[3];
    const float* fs    = (const float*)d_in[4];
    const float* n1w   = (const float*)d_in[5];
    const float* n2w   = (const float*)d_in[6];
    const float* n3w   = (const float*)d_in[7];
    const float* Wqkv  = (const float*)d_in[8];
    const float* Wsa   = (const float*)d_in[9];
    const float* Wq    = (const float*)d_in[10];
    const float* Wkv   = (const float*)d_in[11];
    const float* Wca   = (const float*)d_in[12];
    const float* Wg    = (const float*)d_in[13];
    const float* Wu    = (const float*)d_in[14];
    const float* Wd    = (const float*)d_in[15];
    const float* aW    = (const float*)d_in[16];
    const float* ab    = (const float*)d_in[17];

    char* ws = (char*)d_ws;
    size_t off = 0;
    auto alloc = [&](size_t bytes) -> char* {
        char* p = ws + off;
        off += (bytes + 255) & ~(size_t)255;
        return p;
    };
    bf16*  wqkv_b = (bf16*)alloc(3145728ull * 2);
    bf16*  wsa_b  = (bf16*)alloc(1048576ull * 2);
    bf16*  wq_b   = (bf16*)alloc(1048576ull * 2);
    bf16*  wkv_b  = (bf16*)alloc(1572864ull * 2);
    bf16*  wca_b  = (bf16*)alloc(1048576ull * 2);
    bf16*  wg_b   = (bf16*)alloc(4194304ull * 2);
    bf16*  wu_b   = (bf16*)alloc(4194304ull * 2);
    bf16*  wd_b   = (bf16*)alloc(4194304ull * 2);
    bf16*  aud_b  = (bf16*)alloc(786432ull * 2);
    float* siluT  = (float*)alloc(2048ull * 4);
    float* mods   = (float*)alloc(12288ull * 4);
    bf16*  act_b  = (bf16*)alloc(16777216ull * 2);
    float* bufA   = (float*)alloc(16777216ull * 4);
    float* bufB   = (float*)alloc(16777216ull * 4);
    bf16*  qh     = (bf16*)alloc(4194304ull * 2);
    bf16*  kh     = (bf16*)alloc(4194304ull * 2);
    bf16*  vh     = (bf16*)alloc(4194304ull * 2);
    float* x1     = (float*)alloc(4194304ull * 4);

    // weights: convert + transpose to [N][K] bf16 for the GEMM B operand
    auto cvtT = [&](const float* src, bf16* dst, int K, int N) {
        int n = K * N;
        k_cvt_t<<<dim3((unsigned)((n + 255) / 256)), dim3(256), 0, stream>>>(src, dst, K, N);
    };
    cvtT(Wqkv, wqkv_b, 1024, 3072);
    cvtT(Wsa,  wsa_b,  1024, 1024);
    cvtT(Wq,   wq_b,   1024, 1024);
    cvtT(Wkv,  wkv_b,  768,  2048);
    cvtT(Wca,  wca_b,  1024, 1024);
    cvtT(Wg,   wg_b,   1024, 4096);
    cvtT(Wu,   wu_b,   1024, 4096);
    cvtT(Wd,   wd_b,   4096, 1024);
    k_f32_to_bf16<<<3072, 256, 0, stream>>>(audio, aud_b, 786432);

    // adaLN modulation
    k_silu<<<8, 256, 0, stream>>>(t_mod, siluT, 2048);
    k_adaln<<<48, 256, 0, stream>>>(siluT, aW, ab, mods);

    // ---- self-attention ----
    // mods: shift_sa=0, scale_sa=1024, gate_sa=2048, shift_mlp=3072, scale_mlp=4096, gate_mlp=5120
    k_rmsnorm_mod<<<4096, 256, 0, stream>>>(x, n1w, mods, 1024, 0, act_b);
    k_gemm_bf16<<<dim3(24, 32), 256, 0, stream>>>(act_b, wqkv_b, bufA, 4096, 3072, 1024);
    k_rope_pack<<<8192, 256, 0, stream>>>(bufA, fc, fs, qh, kh, vh);
    k_flash<<<dim3(32, 32), 128, 0, stream>>>(qh, kh, vh, act_b, 2048, 2048, 0.125f);
    k_gemm_bf16<<<dim3(8, 32), 256, 0, stream>>>(act_b, wsa_b, bufA, 4096, 1024, 1024);
    k_residual<<<16384, 256, 0, stream>>>(x, bufA, mods, 2048, x1, 4194304);

    // ---- cross-attention ----
    k_rmsnorm_mod<<<4096, 256, 0, stream>>>(x1, n2w, mods, -1, -1, act_b);
    k_gemm_bf16<<<dim3(8, 32), 256, 0, stream>>>(act_b, wq_b, bufA, 4096, 1024, 1024);
    k_gemm_bf16<<<dim3(16, 8), 256, 0, stream>>>(aud_b, wkv_b, bufB, 1024, 2048, 768);
    k_pack_q<<<16384, 256, 0, stream>>>(bufA, qh);
    k_pack_kv<<<4096, 256, 0, stream>>>(bufB, kh, vh);
    k_flash<<<dim3(32, 32), 128, 0, stream>>>(qh, kh, vh, act_b, 2048, 512, 0.125f);
    k_gemm_bf16<<<dim3(8, 32), 256, 0, stream>>>(act_b, wca_b, bufA, 4096, 1024, 1024);
    k_residual<<<16384, 256, 0, stream>>>(x1, bufA, mods, -1, x1, 4194304);

    // ---- MLP (SwiGLU) ----
    k_rmsnorm_mod<<<4096, 256, 0, stream>>>(x1, n3w, mods, 4096, 3072, act_b);
    k_gemm_bf16<<<dim3(32, 32), 256, 0, stream>>>(act_b, wg_b, bufA, 4096, 4096, 1024);
    k_gemm_bf16<<<dim3(32, 32), 256, 0, stream>>>(act_b, wu_b, bufB, 4096, 4096, 1024);
    k_swiglu<<<65536, 256, 0, stream>>>(bufA, bufB, act_b, 16777216);
    k_gemm_bf16<<<dim3(8, 32), 256, 0, stream>>>(act_b, wd_b, bufA, 4096, 1024, 4096);
    k_residual<<<16384, 256, 0, stream>>>(x1, bufA, mods, 5120, (float*)d_out, 4194304);
}
